// MLPAttention_6090263625762
// MI455X (gfx1250) — compile-verified, matrix operand order checked by reading, not run on hardware
//
#include <hip/hip_runtime.h>
#include <hip/hip_bf16.h>

// Problem dims (fixed by the reference)
#define B_   32
#define LQ_  1024
#define LK_  1024
#define D_   512
#define H_   512
#define DV_  512
#define MASKV (-1.0e6f)

typedef __bf16 v16bf __attribute__((ext_vector_type(16)));
typedef float  v8f   __attribute__((ext_vector_type(8)));

__device__ __forceinline__ void split_bf16(float x, __bf16& hi, __bf16& lo) {
    hi = (__bf16)x;                 // RNE to bf16
    lo = (__bf16)(x - (float)hi);   // residual, ~2^-16 relative
}

// -----------------------------------------------------------------------------
// Stage 0: split W_k into hi/lo bf16 once, stored pre-swizzled in the WMMA
// B-operand per-lane register layout: [ntile][kstep][lane][16 bf16] so the
// GEMM kernel loads each B tile as two B128 loads per lane, zero conversion.
// -----------------------------------------------------------------------------
__global__ __launch_bounds__(32)
void prep_wk_kernel(const float* __restrict__ Wk,
                    __bf16* __restrict__ wkhi, __bf16* __restrict__ wklo) {
    const int ntile = blockIdx.x;          // 0..31  (H/16)
    const int kstep = blockIdx.y;          // 0..15  (D/32)
    const int lane  = threadIdx.x;         // 0..31
    const int n     = ntile * 16 + (lane & 15);
    const int koff  = (lane >> 4) * 8;

    union { __bf16 h[16]; uint4 q[2]; } uh, ul;
    #pragma unroll
    for (int j = 0; j < 8; ++j) {
        const int kk = kstep * 32 + ((j < 4) ? 0 : 16) + koff + 2 * (j & 3);
        __bf16 h, l;
        split_bf16(Wk[(size_t)kk * H_ + n],       h, l); uh.h[2*j]   = h; ul.h[2*j]   = l;
        split_bf16(Wk[(size_t)(kk + 1) * H_ + n], h, l); uh.h[2*j+1] = h; ul.h[2*j+1] = l;
    }
    const size_t idx = ((size_t)(ntile * 16 + kstep) * 32 + lane) * 2;
    ((uint4*)wkhi)[idx]     = uh.q[0];
    ((uint4*)wkhi)[idx + 1] = uh.q[1];
    ((uint4*)wklo)[idx]     = ul.q[0];
    ((uint4*)wklo)[idx + 1] = ul.q[1];
}

// -----------------------------------------------------------------------------
// Stage 1: ks[b,l] = tanh(key[b,l,:] @ W_k) . v_w   (fused; split-bf16 WMMA)
// One wave owns a 16-row tile; 8 N-tiles per k-step (A split amortized 8x).
// -----------------------------------------------------------------------------
__global__ __launch_bounds__(128)
void ks_kernel(const float* __restrict__ key,
               const __bf16* __restrict__ wkhi, const __bf16* __restrict__ wklo,
               const float* __restrict__ vw, float* __restrict__ ks) {
    const int lane = threadIdx.x & 31;
    const int wave = threadIdx.x >> 5;
    const int b    = blockIdx.y;
    const int row0 = (blockIdx.x * 4 + wave) * 16;
    const int m    = lane & 15;
    const int koff = (lane >> 4) * 8;

    const float* arow = key + ((size_t)b * LK_ + row0 + m) * D_;
    const uint4* bh4  = (const uint4*)wkhi;
    const uint4* bl4  = (const uint4*)wklo;

    float rowsum[8];
    #pragma unroll
    for (int i = 0; i < 8; ++i) rowsum[i] = 0.0f;

    // H = 512 -> 4 groups of 8 N-tiles (16 cols each)
    for (int ng = 0; ng < H_ / 128; ++ng) {
        v8f acc[8];
        #pragma unroll
        for (int t = 0; t < 8; ++t)
            #pragma unroll
            for (int i = 0; i < 8; ++i) acc[t][i] = 0.0f;

        for (int kstep = 0; kstep < D_ / 32; ++kstep) {
            // ---- A tile (16x32 of key rows): split fp32 -> bf16 hi/lo ----
            v16bf ahi, alo;
            #pragma unroll
            for (int j = 0; j < 8; ++j) {
                const int kk = kstep * 32 + ((j < 4) ? 0 : 16) + koff + 2 * (j & 3);
                __bf16 h, l;
                split_bf16(arow[kk],     h, l); ahi[2*j]   = h; alo[2*j]   = l;
                split_bf16(arow[kk + 1], h, l); ahi[2*j+1] = h; alo[2*j+1] = l;
            }
            // ---- 8 pre-split B tiles; 3 WMMAs each (hi*hi + hi*lo + lo*hi) ----
            #pragma unroll
            for (int t = 0; t < 8; ++t) {
                const int nt = ng * 8 + t;
                const size_t bidx = ((size_t)(nt * 16 + kstep) * 32 + lane) * 2;
                union { uint4 q[2]; v16bf v; } ubh, ubl;
                ubh.q[0] = bh4[bidx]; ubh.q[1] = bh4[bidx + 1];
                ubl.q[0] = bl4[bidx]; ubl.q[1] = bl4[bidx + 1];
                acc[t] = __builtin_amdgcn_wmma_f32_16x16x32_bf16(
                             false, ahi, false, ubh.v, (short)0, acc[t], false, false);
                acc[t] = __builtin_amdgcn_wmma_f32_16x16x32_bf16(
                             false, ahi, false, ubl.v, (short)0, acc[t], false, false);
                acc[t] = __builtin_amdgcn_wmma_f32_16x16x32_bf16(
                             false, alo, false, ubh.v, (short)0, acc[t], false, false);
            }
        }
        // tanh + dot with v_w chunk. C layout: lane has N = m, M = i + 8*(lane>=16)
        #pragma unroll
        for (int t = 0; t < 8; ++t) {
            const int n = (ng * 8 + t) * 16 + m;
            const float vwn = vw[n];
            #pragma unroll
            for (int i = 0; i < 8; ++i)
                rowsum[i] += tanhf(acc[t][i]) * vwn;
        }
    }

    // Reduce over the 16 lanes of each half-wave (sum over N)
    #pragma unroll
    for (int off = 1; off < 16; off <<= 1) {
        #pragma unroll
        for (int i = 0; i < 8; ++i)
            rowsum[i] += __shfl_xor(rowsum[i], off, 32);
    }
    if (m == 0) {
        const int mbase = (lane >> 4) * 8;   // lane0 -> rows 0..7, lane16 -> rows 8..15
        #pragma unroll
        for (int i = 0; i < 8; ++i)
            ks[(size_t)b * LK_ + row0 + mbase + i] = rowsum[i];
    }
}

// -----------------------------------------------------------------------------
// Stage 2: w[b,:] = softmax over valid k of ks[b,:]; 0 for masked (exact in fp32)
// -----------------------------------------------------------------------------
__global__ __launch_bounds__(256)
void softmax_kernel(const float* __restrict__ ks, const int* __restrict__ vlen,
                    float* __restrict__ w) {
    __shared__ float red[256];
    const int b = blockIdx.x;
    const int t = threadIdx.x;
    const int n = vlen[b];

    float vals[4];
    float mx = -3.4e38f;
    #pragma unroll
    for (int i = 0; i < 4; ++i) {
        const int k = t + i * 256;
        const float s = (k < n) ? ks[(size_t)b * LK_ + k] : MASKV;
        vals[i] = s;
        mx = fmaxf(mx, s);
    }
    red[t] = mx; __syncthreads();
    for (int off = 128; off > 0; off >>= 1) {
        if (t < off) red[t] = fmaxf(red[t], red[t + off]);
        __syncthreads();
    }
    mx = red[0]; __syncthreads();

    float e[4];
    float sum = 0.0f;
    #pragma unroll
    for (int i = 0; i < 4; ++i) { e[i] = __expf(vals[i] - mx); sum += e[i]; }
    red[t] = sum; __syncthreads();
    for (int off = 128; off > 0; off >>= 1) {
        if (t < off) red[t] = red[t] + red[t + off];
        __syncthreads();
    }
    const float inv = 1.0f / red[0];
    #pragma unroll
    for (int i = 0; i < 4; ++i) {
        const int k = t + i * 256;
        w[(size_t)b * LK_ + k] = (k < n) ? e[i] * inv : 0.0f;
    }
}

// -----------------------------------------------------------------------------
// Stage 3: pctx[b,kc,:] = sum_{k in chunk kc} w[b,k] * value[b,k,:]
// Deterministic fixed-order partials (no atomics).
// -----------------------------------------------------------------------------
__global__ __launch_bounds__(256)
void ctx_partial_kernel(const float* __restrict__ w, const float* __restrict__ value,
                        float* __restrict__ pctx) {
    const int b  = blockIdx.z;
    const int vt = blockIdx.y;                   // 0..1
    const int kc = blockIdx.x;                   // 0..7
    const int v  = vt * 256 + threadIdx.x;
    const int k0 = kc * (LK_ / 8);

    float acc = 0.0f;
    for (int k = k0; k < k0 + LK_ / 8; ++k) {
        const float wk = w[(size_t)b * LK_ + k];
        acc += wk * value[((size_t)b * LK_ + k) * DV_ + v];
    }
    pctx[((size_t)b * 8 + kc) * DV_ + v] = acc;
}

// -----------------------------------------------------------------------------
// Stage 4: out[b,q,:] = sum_kc pctx[b,kc,:]  (row broadcast over all q)
// -----------------------------------------------------------------------------
__global__ __launch_bounds__(256)
void bcast_kernel(const float* __restrict__ pctx, float* __restrict__ out) {
    const int b  = blockIdx.y;
    const int q0 = blockIdx.x * 16;
    const int t  = threadIdx.x;

    float c0 = 0.0f, c1 = 0.0f;
    #pragma unroll
    for (int kc = 0; kc < 8; ++kc) {
        c0 += pctx[((size_t)b * 8 + kc) * DV_ + t];
        c1 += pctx[((size_t)b * 8 + kc) * DV_ + t + 256];
    }
    for (int q = q0; q < q0 + 16; ++q) {
        float* o = out + ((size_t)b * LQ_ + q) * DV_;
        o[t]       = c0;
        o[t + 256] = c1;
    }
}

// -----------------------------------------------------------------------------
// Inputs (setup_inputs order): 0=query, 1=key, 2=value, 3=valid_length(int32),
//                              4=W_q, 5=W_k, 6=v_w
// query/W_q are provably unused (softmax shift invariance; masked lanes exactly 0).
// -----------------------------------------------------------------------------
extern "C" void kernel_launch(void* const* d_in, const int* in_sizes, int n_in,
                              void* d_out, int out_size, void* d_ws, size_t ws_size,
                              hipStream_t stream) {
    const float* key   = (const float*)d_in[1];
    const float* value = (const float*)d_in[2];
    const int*   vlen  = (const int*)  d_in[3];
    const float* Wk    = (const float*)d_in[5];
    const float* vw    = (const float*)d_in[6];
    float* out = (float*)d_out;

    float*  ks   = (float*)d_ws;                       // B*LK floats   (128 KiB)
    float*  w    = ks + (size_t)B_ * LK_;              // B*LK floats   (128 KiB)
    float*  pctx = w  + (size_t)B_ * LK_;              // B*8*DV floats (512 KiB)
    __bf16* wkhi = (__bf16*)(pctx + (size_t)B_ * 8 * DV_);   // D*H bf16 (512 KiB)
    __bf16* wklo = wkhi + (size_t)D_ * H_;                   // D*H bf16 (512 KiB)

    prep_wk_kernel<<<dim3(H_ / 16, D_ / 32), 32, 0, stream>>>(Wk, wkhi, wklo);
    ks_kernel<<<dim3(LK_ / 64, B_), 128, 0, stream>>>(key, wkhi, wklo, vw, ks);
    softmax_kernel<<<dim3(B_), 256, 0, stream>>>(ks, vlen, w);
    ctx_partial_kernel<<<dim3(8, 2, B_), 256, 0, stream>>>(w, value, pctx);
    bcast_kernel<<<dim3(LQ_ / 16, B_), 256, 0, stream>>>(pctx, out);
}
